// EmbeddingNet_4612794876591
// MI455X (gfx1250) — matrix operand order repaired and checked by reading, not compiled, not run on hardware
//
#include <hip/hip_runtime.h>
#include <hip/hip_bf16.h>

typedef _Float16 half_t;
typedef __attribute__((ext_vector_type(16))) _Float16 v16h;
typedef __attribute__((ext_vector_type(8)))  _Float16 v8h;
typedef __attribute__((ext_vector_type(8)))  float    v8f;
typedef __attribute__((ext_vector_type(4)))  float    v4f;

#define Nn   80000
#define Ee   1000000
#define FIN  32
#define DIMD 128
#define Bb   64

__device__ __forceinline__ float lrelu_f(float x) { return x > 0.f ? x : 0.01f * x; }
__device__ __forceinline__ float sigm_f(float x)  { return 1.f / (1.f + expf(-x)); }

// order-preserving float -> uint key for atomic segment-max
__device__ __forceinline__ unsigned ordkey(float f) {
    int b = __float_as_int(f);
    unsigned u = (unsigned)b;
    return u ^ ((b < 0) ? 0xFFFFFFFFu : 0x80000000u);
}
__device__ __forceinline__ float fromkey(unsigned u) {
    unsigned b = (u & 0x80000000u) ? (u ^ 0x80000000u) : ~u;
    return __int_as_float((int)b);
}

// ---------------------------------------------------------------- utilities
__global__ void zero_kernel(float* p, int n) {
    int i = blockIdx.x * blockDim.x + threadIdx.x;
    if (i < n) p[i] = 0.f;
}

// src [K][128] row-major f32 -> dst [128][K] f16 (transposed, for B fragments)
__global__ void transpose_to_f16(const float* __restrict__ src, half_t* __restrict__ dst, int K) {
    int idx = blockIdx.x * blockDim.x + threadIdx.x;
    if (idx < K * 128) {
        int k = idx / 128, n = idx % 128;
        dst[n * K + k] = (half_t)src[idx];
    }
}

// ---------------------------------------------------------------- edge MLP
// w[l][e] = elu( lrelu(ea @ ew1[l] + eb1[l]) @ ew2[l] + eb2[l] )
__global__ void edge_mlp_kernel(const float* __restrict__ ea,
                                const float* __restrict__ ew1, const float* __restrict__ eb1,
                                const float* __restrict__ ew2, const float* __restrict__ eb2,
                                float* __restrict__ ewout) {
    int e = blockIdx.x * blockDim.x + threadIdx.x;
    if (e >= Ee) return;
    float a0 = ea[e * 2 + 0], a1 = ea[e * 2 + 1];
#pragma unroll
    for (int l = 0; l < 4; ++l) {
        const float* W1 = ew1 + l * 16;   // [2][8]
        float acc = eb2[l];
#pragma unroll
        for (int j = 0; j < 8; ++j) {
            float hj = a0 * W1[j] + a1 * W1[8 + j] + eb1[l * 8 + j];
            hj = lrelu_f(hj);
            acc += hj * ew2[l * 8 + j];
        }
        ewout[(size_t)l * Ee + e] = acc > 0.f ? acc : expm1f(acc);
    }
}

// ---------------------------------------------------------------- aggregation
__global__ void init_agg_kernel(const float* __restrict__ hin, float* __restrict__ agg,
                                const float* __restrict__ eps, int l, int total) {
    int i = blockIdx.x * blockDim.x + threadIdx.x;
    if (i < total) agg[i] = (1.f + eps[l]) * hin[i];
}

// one wave per edge: agg[dst] += h[src] * w[e]
__global__ void scatter_kernel(const float* __restrict__ hin, const float* __restrict__ ewl,
                               const int* __restrict__ ei, float* __restrict__ agg, int F) {
    int w = (blockIdx.x * blockDim.x + threadIdx.x) >> 5;
    int lane = threadIdx.x & 31;
    if (w >= Ee) return;
    int src = ei[w];
    int dst = ei[Ee + w];
    float wv = ewl[w];
    for (int f = lane; f < F; f += 32)
        atomicAdd(&agg[(size_t)dst * F + f], hin[(size_t)src * F + f] * wv);
}

// ---------------------------------------------------------------- WMMA GEMMs
// t = lrelu(A[f32, M x K] @ Wt^T + bias) stored f16.  Wt is [128][K] f16.
__global__ void gemm1_kernel(const float* __restrict__ A, const half_t* __restrict__ Wt,
                             const float* __restrict__ bias, half_t* __restrict__ out, int K) {
    const int lane = threadIdx.x & 31;
    const int wave = threadIdx.x >> 5;                 // column tile 0..7
    const int row  = blockIdx.x * 16 + (lane & 15);
    const int col  = wave * 16 + (lane & 15);
    const int kbA  = (lane < 16) ? 0 : 8;
    const int kbB  = (lane < 16) ? 0 : 8;
    v8f c;
    float bv = bias[col];
#pragma unroll
    for (int v = 0; v < 8; ++v) c[v] = bv;
    for (int kk = 0; kk < K; kk += 32) {
        v16h a, b;
        const v4f* pa = (const v4f*)(A + (size_t)row * K + kk + kbA);
        v4f x0 = pa[0], x1 = pa[1];
        const v4f* pa2 = (const v4f*)(A + (size_t)row * K + kk + kbA + 16);
        v4f x2 = pa2[0], x3 = pa2[1];
#pragma unroll
        for (int j = 0; j < 4; ++j) {
            a[j]      = (half_t)x0[j];
            a[4 + j]  = (half_t)x1[j];
            a[8 + j]  = (half_t)x2[j];
            a[12 + j] = (half_t)x3[j];
        }
        const v8h* pb = (const v8h*)(Wt + (size_t)col * K + kk + kbB);
        v8h y0 = pb[0], y1 = pb[2];
#pragma unroll
        for (int j = 0; j < 8; ++j) { b[j] = y0[j]; b[8 + j] = y1[j]; }
        c = __builtin_amdgcn_wmma_f32_16x16x32_f16(false, a, false, b, (short)0, c, false, false);
    }
    const int rbase = blockIdx.x * 16 + ((lane >> 4) << 3);
#pragma unroll
    for (int v = 0; v < 8; ++v)
        out[(size_t)(rbase + v) * DIMD + col] = (half_t)lrelu_f(c[v]);
}

// u = lrelu(A16[f16, M x 128] @ Wt^T + bias) stored f32.
__global__ void gemm2_kernel(const half_t* __restrict__ A16, const half_t* __restrict__ Wt,
                             const float* __restrict__ bias, float* __restrict__ out) {
    const int K = DIMD;
    const int lane = threadIdx.x & 31;
    const int wave = threadIdx.x >> 5;
    const int row  = blockIdx.x * 16 + (lane & 15);
    const int col  = wave * 16 + (lane & 15);
    const int kbA  = (lane < 16) ? 0 : 8;
    const int kbB  = (lane < 16) ? 0 : 8;
    v8f c;
    float bv = bias[col];
#pragma unroll
    for (int v = 0; v < 8; ++v) c[v] = bv;
    for (int kk = 0; kk < K; kk += 32) {
        v16h a, b;
        const v8h* pa = (const v8h*)(A16 + (size_t)row * K + kk + kbA);
        v8h x0 = pa[0];
        const v8h* pa2 = (const v8h*)(A16 + (size_t)row * K + kk + kbA + 16);
        v8h x1 = pa2[0];
#pragma unroll
        for (int j = 0; j < 8; ++j) { a[j] = x0[j]; a[8 + j] = x1[j]; }
        const v8h* pb = (const v8h*)(Wt + (size_t)col * K + kk + kbB);
        v8h y0 = pb[0], y1 = pb[2];
#pragma unroll
        for (int j = 0; j < 8; ++j) { b[j] = y0[j]; b[8 + j] = y1[j]; }
        c = __builtin_amdgcn_wmma_f32_16x16x32_f16(false, a, false, b, (short)0, c, false, false);
    }
    const int rbase = blockIdx.x * 16 + ((lane >> 4) << 3);
#pragma unroll
    for (int v = 0; v < 8; ++v)
        out[(size_t)(rbase + v) * DIMD + col] = lrelu_f(c[v]);
}

// ---------------------------------------------------------------- BatchNorm
__global__ void bn_stats_kernel(const float* __restrict__ u, float* __restrict__ stats) {
    int col = threadIdx.x;                         // 128 threads
    int rows_per = (Nn + gridDim.x - 1) / gridDim.x;
    int r0 = blockIdx.x * rows_per;
    int r1 = min(r0 + rows_per, Nn);
    float s = 0.f, s2 = 0.f;
    for (int r = r0; r < r1; ++r) {
        float v = u[(size_t)r * DIMD + col];
        s += v; s2 += v * v;
    }
    atomicAdd(&stats[col], s);
    atomicAdd(&stats[DIMD + col], s2);
}

__global__ void bn_apply_kernel(const float* __restrict__ u, const float* __restrict__ stats,
                                const float* __restrict__ g, const float* __restrict__ bt,
                                float* __restrict__ hout, int total) {
    int i = blockIdx.x * blockDim.x + threadIdx.x;
    if (i >= total) return;
    int col = i & (DIMD - 1);
    float mean = stats[col] * (1.f / Nn);
    float var  = stats[DIMD + col] * (1.f / Nn) - mean * mean;
    hout[i] = g[col] * (u[i] - mean) * rsqrtf(var + 1e-5f) + bt[col];
}

// ---------------------------------------------------------------- Set2Set
__global__ void lstm_kernel(const float* __restrict__ qs,
                            float* __restrict__ h0, float* __restrict__ c0,
                            float* __restrict__ h1, float* __restrict__ c1,
                            const float* __restrict__ Wih0, const float* __restrict__ Whh0,
                            const float* __restrict__ bih0, const float* __restrict__ bhh0,
                            const float* __restrict__ Wih1, const float* __restrict__ Whh1,
                            const float* __restrict__ bih1, const float* __restrict__ bhh1,
                            float* __restrict__ q, float* __restrict__ r,
                            float* __restrict__ s, unsigned* __restrict__ mkey) {
    __shared__ float gate[512];
    __shared__ float hnew[DIMD];
    const int b = blockIdx.x, t = threadIdx.x;
    // layer 0: input = q_star [256]
    {
        float acc = bih0[t] + bhh0[t];
        const float* wi = Wih0 + (size_t)t * 256;
        const float* qr = qs + (size_t)b * 256;
        for (int k = 0; k < 256; ++k) acc += qr[k] * wi[k];
        const float* wh = Whh0 + (size_t)t * DIMD;
        const float* hr = h0 + (size_t)b * DIMD;
        for (int k = 0; k < DIMD; ++k) acc += hr[k] * wh[k];
        gate[t] = acc;
    }
    __syncthreads();
    if (t < DIMD) {
        float ig = sigm_f(gate[t]);
        float fg = sigm_f(gate[DIMD + t]);
        float gg = tanhf(gate[2 * DIMD + t]);
        float og = sigm_f(gate[3 * DIMD + t]);
        float c = fg * c0[b * DIMD + t] + ig * gg;
        float h = og * tanhf(c);
        c0[b * DIMD + t] = c;
        h0[b * DIMD + t] = h;
        hnew[t] = h;
    }
    __syncthreads();
    // layer 1: input = new h0 [128]
    {
        float acc = bih1[t] + bhh1[t];
        const float* wi = Wih1 + (size_t)t * DIMD;
        for (int k = 0; k < DIMD; ++k) acc += hnew[k] * wi[k];
        const float* wh = Whh1 + (size_t)t * DIMD;
        const float* hr = h1 + (size_t)b * DIMD;
        for (int k = 0; k < DIMD; ++k) acc += hr[k] * wh[k];
        gate[t] = acc;
    }
    __syncthreads();
    if (t < DIMD) {
        float ig = sigm_f(gate[t]);
        float fg = sigm_f(gate[DIMD + t]);
        float gg = tanhf(gate[2 * DIMD + t]);
        float og = sigm_f(gate[3 * DIMD + t]);
        float c = fg * c1[b * DIMD + t] + ig * gg;
        float h = og * tanhf(c);
        c1[b * DIMD + t] = c;
        h1[b * DIMD + t] = h;
        q[b * DIMD + t] = h;
        r[b * DIMD + t] = 0.f;          // reset attention readout accumulators
    }
    if (t == 0) { s[b] = 0.f; mkey[b] = 0u; }
}

// e[n] = dot(h[n], q[batch[n]]); segment-max via ordered-uint atomicMax
__global__ void score_kernel(const float* __restrict__ h, const float* __restrict__ q,
                             const int* __restrict__ batch, float* __restrict__ e,
                             unsigned* __restrict__ mkey) {
    int w = (blockIdx.x * blockDim.x + threadIdx.x) >> 5;
    int lane = threadIdx.x & 31;
    if (w >= Nn) return;
    int b = batch[w];
    float acc = 0.f;
#pragma unroll
    for (int j = 0; j < 4; ++j)
        acc += h[(size_t)w * DIMD + lane + 32 * j] * q[b * DIMD + lane + 32 * j];
    for (int off = 16; off > 0; off >>= 1) acc += __shfl_down(acc, off);
    if (lane == 0) {
        e[w] = acc;
        atomicMax(&mkey[b], ordkey(acc));
    }
}

__global__ void exp_kernel(const float* __restrict__ e, const int* __restrict__ batch,
                           const unsigned* __restrict__ mkey, float* __restrict__ a,
                           float* __restrict__ s) {
    int n = blockIdx.x * blockDim.x + threadIdx.x;
    if (n >= Nn) return;
    int b = batch[n];
    float av = expf(e[n] - fromkey(mkey[b]));
    a[n] = av;
    atomicAdd(&s[b], av);
}

__global__ void accum_r_kernel(const float* __restrict__ h, const float* __restrict__ a,
                               const float* __restrict__ s, const int* __restrict__ batch,
                               float* __restrict__ r) {
    int w = (blockIdx.x * blockDim.x + threadIdx.x) >> 5;
    int lane = threadIdx.x & 31;
    if (w >= Nn) return;
    int b = batch[w];
    float coeff = a[w] / s[b];
#pragma unroll
    for (int j = 0; j < 4; ++j)
        atomicAdd(&r[b * DIMD + lane + 32 * j], coeff * h[(size_t)w * DIMD + lane + 32 * j]);
}

__global__ void qstar_kernel(const float* __restrict__ q, const float* __restrict__ r,
                             float* __restrict__ qstar, float* __restrict__ out, int final_step) {
    int i = blockIdx.x * blockDim.x + threadIdx.x;
    if (i >= Bb * 256) return;
    int b = i >> 8, k = i & 255;
    float v = (k < DIMD) ? q[b * DIMD + k] : r[b * DIMD + (k - DIMD)];
    qstar[i] = v;
    if (final_step) out[i] = v;
}

// ---------------------------------------------------------------- launcher
extern "C" void kernel_launch(void* const* d_in, const int* in_sizes, int n_in,
                              void* d_out, int out_size, void* d_ws, size_t ws_size,
                              hipStream_t stream) {
    const float* x        = (const float*)d_in[0];
    const int*   ei       = (const int*)d_in[1];
    const float* eattr    = (const float*)d_in[2];
    const int*   batch    = (const int*)d_in[3];
    const float* ew1      = (const float*)d_in[4];
    const float* eb1      = (const float*)d_in[5];
    const float* ew2      = (const float*)d_in[6];
    const float* eb2      = (const float*)d_in[7];
    const float* eps      = (const float*)d_in[8];
    const float* w1_0     = (const float*)d_in[9];
    const float* b1_0     = (const float*)d_in[10];
    const float* w1_rest  = (const float*)d_in[11];
    const float* b1_rest  = (const float*)d_in[12];
    const float* w2_all   = (const float*)d_in[13];
    const float* b2_all   = (const float*)d_in[14];
    const float* bn_g     = (const float*)d_in[15];
    const float* bn_b     = (const float*)d_in[16];
    const float* Wih0     = (const float*)d_in[17];
    const float* Whh0     = (const float*)d_in[18];
    const float* bih0     = (const float*)d_in[19];
    const float* bhh0     = (const float*)d_in[20];
    const float* Wih1     = (const float*)d_in[21];
    const float* Whh1     = (const float*)d_in[22];
    const float* bih1     = (const float*)d_in[23];
    const float* bhh1     = (const float*)d_in[24];

    // workspace bump allocator (256B aligned)
    char*  base = (char*)d_ws;
    size_t off  = 0;
    auto alloc = [&](size_t bytes) -> void* {
        void* p = base + off;
        off = (off + bytes + 255) & ~(size_t)255;
        return p;
    };
    half_t* wt1_0  = (half_t*)alloc((size_t)FIN * DIMD * 2);
    half_t* wt1_r  = (half_t*)alloc((size_t)3 * DIMD * DIMD * 2);
    half_t* wt2    = (half_t*)alloc((size_t)4 * DIMD * DIMD * 2);
    float*  ew     = (float*)alloc((size_t)4 * Ee * 4);
    float*  agg    = (float*)alloc((size_t)Nn * DIMD * 4);   // also reused as u
    half_t* tbuf   = (half_t*)alloc((size_t)Nn * DIMD * 2);
    float*  hbuf   = (float*)alloc((size_t)Nn * DIMD * 4);
    float*  stats  = (float*)alloc(2 * DIMD * 4);
    float*  ebuf   = (float*)alloc((size_t)Nn * 4);
    float*  abuf   = (float*)alloc((size_t)Nn * 4);
    float*  qstar  = (float*)alloc((size_t)Bb * 256 * 4);
    float*  h0     = (float*)alloc((size_t)Bb * DIMD * 4);
    float*  c0     = (float*)alloc((size_t)Bb * DIMD * 4);
    float*  h1     = (float*)alloc((size_t)Bb * DIMD * 4);
    float*  c1     = (float*)alloc((size_t)Bb * DIMD * 4);
    float*  qv     = (float*)alloc((size_t)Bb * DIMD * 4);
    float*  rv     = (float*)alloc((size_t)Bb * DIMD * 4);
    float*  sv     = (float*)alloc((size_t)Bb * 4);
    unsigned* mkey = (unsigned*)alloc((size_t)Bb * 4);
    (void)in_sizes; (void)n_in; (void)out_size; (void)ws_size;

    // 1) weights -> transposed f16
    transpose_to_f16<<<(FIN * DIMD + 255) / 256, 256, 0, stream>>>(w1_0, wt1_0, FIN);
    for (int l = 0; l < 3; ++l)
        transpose_to_f16<<<(DIMD * DIMD + 255) / 256, 256, 0, stream>>>(
            w1_rest + (size_t)l * DIMD * DIMD, wt1_r + (size_t)l * DIMD * DIMD, DIMD);
    for (int l = 0; l < 4; ++l)
        transpose_to_f16<<<(DIMD * DIMD + 255) / 256, 256, 0, stream>>>(
            w2_all + (size_t)l * DIMD * DIMD, wt2 + (size_t)l * DIMD * DIMD, DIMD);

    // 2) per-edge scalar weights for all 4 layers
    edge_mlp_kernel<<<(Ee + 255) / 256, 256, 0, stream>>>(eattr, ew1, eb1, ew2, eb2, ew);

    // 3) four conv + BN layers
    for (int l = 0; l < 4; ++l) {
        int F = (l == 0) ? FIN : DIMD;
        const float* hin = (l == 0) ? x : hbuf;
        zero_kernel<<<1, 256, 0, stream>>>(stats, 2 * DIMD);
        init_agg_kernel<<<((size_t)Nn * F + 255) / 256, 256, 0, stream>>>(hin, agg, eps, l, Nn * F);
        scatter_kernel<<<((size_t)Ee * 32 + 255) / 256, 256, 0, stream>>>(
            hin, ew + (size_t)l * Ee, ei, agg, F);
        const half_t* W1 = (l == 0) ? wt1_0 : wt1_r + (size_t)(l - 1) * DIMD * DIMD;
        const float*  B1 = (l == 0) ? b1_0 : b1_rest + (size_t)(l - 1) * DIMD;
        gemm1_kernel<<<Nn / 16, 256, 0, stream>>>(agg, W1, B1, tbuf, F);
        gemm2_kernel<<<Nn / 16, 256, 0, stream>>>(
            tbuf, wt2 + (size_t)l * DIMD * DIMD, b2_all + (size_t)l * DIMD, agg);
        bn_stats_kernel<<<256, DIMD, 0, stream>>>(agg, stats);
        bn_apply_kernel<<<((size_t)Nn * DIMD + 255) / 256, 256, 0, stream>>>(
            agg, stats, bn_g + (size_t)l * DIMD, bn_b + (size_t)l * DIMD, hbuf, Nn * DIMD);
    }

    // 4) Set2Set: zero recurrent state, then 5 processing steps
    zero_kernel<<<(Bb * 256 + 255) / 256, 256, 0, stream>>>(qstar, Bb * 256);
    zero_kernel<<<(Bb * DIMD + 255) / 256, 256, 0, stream>>>(h0, Bb * DIMD);
    zero_kernel<<<(Bb * DIMD + 255) / 256, 256, 0, stream>>>(c0, Bb * DIMD);
    zero_kernel<<<(Bb * DIMD + 255) / 256, 256, 0, stream>>>(h1, Bb * DIMD);
    zero_kernel<<<(Bb * DIMD + 255) / 256, 256, 0, stream>>>(c1, Bb * DIMD);
    for (int step = 0; step < 5; ++step) {
        lstm_kernel<<<Bb, 512, 0, stream>>>(qstar, h0, c0, h1, c1,
                                            Wih0, Whh0, bih0, bhh0,
                                            Wih1, Whh1, bih1, bhh1,
                                            qv, rv, sv, mkey);
        score_kernel<<<((size_t)Nn * 32 + 255) / 256, 256, 0, stream>>>(hbuf, qv, batch, ebuf, mkey);
        exp_kernel<<<(Nn + 255) / 256, 256, 0, stream>>>(ebuf, batch, mkey, abuf, sv);
        accum_r_kernel<<<((size_t)Nn * 32 + 255) / 256, 256, 0, stream>>>(hbuf, abuf, sv, batch, rv);
        qstar_kernel<<<(Bb * 256 + 255) / 256, 256, 0, stream>>>(
            qv, rv, qstar, (float*)d_out, step == 4 ? 1 : 0);
    }
}
